// TensorBase_58969900974424
// MI455X (gfx1250) — compile-verified, matrix operand order checked by reading, not run on hardware
//
#include <hip/hip_runtime.h>
#include <hip/hip_bf16.h>
#include <math.h>

typedef __attribute__((ext_vector_type(16))) _Float16 v16h;
typedef __attribute__((ext_vector_type(8)))  _Float16 half8;
typedef __attribute__((ext_vector_type(2)))  _Float16 half2v;
typedef __attribute__((ext_vector_type(8)))  float    v8f;

#define APP_DIM 27
#define NCH   30       // 27 app channels + 3 view channels
#define KIN   390      // 30 raw + 30*12 PE
#define KPAD  416      // 13 * 32
#define FEAT  128
#define NS    128
#define TILE_M 16
#define NTILES (NS / TILE_M)

// ---- K-permutation -------------------------------------------------------
// Our mlp_in layout (compute-friendly):
//   c in [0,30)   : raw app(27) + viewdirs(3)
//   c in [30,390) : j=c-30; ch=j/12; t=j%12; t<6 -> sin(2^t * x_ch)
//                                            t>=6 -> cos(2^(t-6) * x_ch)
//   c in [390,416): zero pad
// Reference layout of W1 rows: [app 0..26 | view 27..29 | PEapp sin 30..191 |
//   PEapp cos 192..353 | PEview sin 354..371 | PEview cos 372..389].
// orig_k1() maps our column -> reference W1 row (applied at fragment load).
__device__ __forceinline__ int orig_k1(int c) {
  if (c < 30) return c;
  int j = c - 30;
  int ch = j / 12, t = j - ch * 12;
  int isCos = (t >= 6) ? 1 : 0;
  int f = t - 6 * isCos;
  return (ch < APP_DIM) ? (30 + isCos * 162 + ch * 6 + f)
                        : (354 + isCos * 18 + (ch - APP_DIM) * 6 + f);
}

// CDNA5 WMMA B-fragment (32x16 f16) gathered from row-major fp32 W[k][col].
// lanes 0-15: K = kb+{0..7,16..23}; lanes 16-31: K = kb+{8..15,24..31}.
__device__ __forceinline__ v16h load_b_f32(const float* __restrict__ W, int ldw,
                                           int krows, int col, int ncols,
                                           int kb, int kHalf, bool remapW1) {
  const int cc = (col < ncols) ? col : 0;
  v16h b;
#pragma unroll
  for (int h = 0; h < 16; ++h) {
    int k = kb + ((h < 8) ? h : h + 8) + kHalf * 8;
    int kk = (k < krows) ? (remapW1 ? orig_k1(k) : k) : 0;
    float v = W[kk * ldw + cc];
    if (k >= krows || col >= ncols) v = 0.0f;
    b[h] = (_Float16)v;
  }
  return b;
}

// CDNA5 WMMA A-fragment (16x32 f16) from row-major f16 LDS:
// two contiguous 16B runs per lane -> 2x ds_load_b128.
__device__ __forceinline__ v16h load_a_lds(const _Float16* base, int ldh,
                                           int m, int kb, int kHalf) {
  const _Float16* p = base + m * ldh + kb + kHalf * 8;
  half8 lo = *(const half8*)p;
  half8 hi = *(const half8*)(p + 16);
  v16h a;
#pragma unroll
  for (int i = 0; i < 8; ++i) { a[i] = lo[i]; a[i + 8] = hi[i]; }
  return a;
}

__global__ __launch_bounds__(256)
void nerf_fused_kernel(const float* __restrict__ sigf,
                       const float* __restrict__ app,
                       const float* __restrict__ vdir,
                       const float* __restrict__ dists,
                       const float* __restrict__ W1, const float* __restrict__ b1,
                       const float* __restrict__ W2, const float* __restrict__ b2,
                       const float* __restrict__ W3, const float* __restrict__ b3,
                       float* __restrict__ out) {
  __shared__ __align__(16) _Float16 s_in[TILE_M * KPAD];   // 13 KB
  __shared__ __align__(16) _Float16 s_h1[TILE_M * FEAT];   // 4 KB
  __shared__ __align__(16) _Float16 s_h2[TILE_M * FEAT];   // 4 KB
  __shared__ float s_raw[TILE_M * 32];                     // raw app+view
  __shared__ float s_w[TILE_M];                            // render weights
  __shared__ float s_T;                                    // running transmittance

  const int tid   = threadIdx.x;
  const int lane  = tid & 31;
  const int wv    = __builtin_amdgcn_readfirstlane(tid >> 5); // uniform wave id
  const int n     = lane & 15;
  const int kHalf = lane >> 4;
  const int ray   = blockIdx.x;
  const long rbase = (long)ray * NS;

  // ---- one-time: swizzle W1/W2/W3 fragments into registers (K-permuted W1) --
  v16h w1f[13], w2f[4], w3f[4];
#pragma unroll
  for (int kt = 0; kt < 13; ++kt)
    w1f[kt] = load_b_f32(W1, FEAT, KIN, 16 * wv + n, FEAT, kt * 32, kHalf, true);
#pragma unroll
  for (int kt = 0; kt < 4; ++kt)
    w2f[kt] = load_b_f32(W2, FEAT, FEAT, 16 * wv + n, FEAT, kt * 32, kHalf, false);
#pragma unroll
  for (int kt = 0; kt < 4; ++kt)
    w3f[kt] = load_b_f32(W3, 3, FEAT, n, 3, kt * 32, kHalf, false);

  const float bv1 = b1[16 * wv + n];
  const float bv2 = b2[16 * wv + n];
  const float bv3 = (n < 3) ? b3[n] : 0.0f;

  if (tid == 0) s_T = 1.0f;
  // zero K-pad columns 390..415 once (never overwritten)
  for (int t = tid; t < TILE_M * 32; t += 256) {
    int m = t >> 5, c = 384 + (t & 31);
    if (c >= KIN) s_in[m * KPAD + c] = (_Float16)0.0f;
  }
  float rgb_sum = 0.0f;
  __syncthreads();

#pragma unroll 1
  for (int tile = 0; tile < NTILES; ++tile) {
    // ---- phase 1: stage raw inputs; thread 0 runs transmittance scan ----
    for (int t = tid; t < TILE_M * 32; t += 256) {
      int m = t >> 5, c = t & 31;
      long p = rbase + tile * TILE_M + m;
      float v = 0.0f;
      if (c < APP_DIM)      v = app[p * APP_DIM + c];
      else if (c < NCH)     v = vdir[p * 3 + (c - APP_DIM)];
      s_raw[t] = v;
    }
    if (tid == 0) {
      float T = s_T;
#pragma unroll 1
      for (int i = 0; i < TILE_M; ++i) {
        int s = tile * TILE_M + i;
        long p = rbase + s;
        float a;
        if (s == NS - 1) a = 1.0f;
        else {
          float x  = sigf[p] - 10.0f;
          float sp = (x > 20.0f) ? x : log1pf(__expf(x));
          a = 1.0f - __expf(-sp * dists[p] * 25.0f);
        }
        s_w[i] = a * T;
        T *= (1.0f - a + 1e-10f);
      }
      s_T = T;
    }
    if (tid == 64 && tile + 1 < NTILES)
      __builtin_prefetch(&app[(rbase + (long)(tile + 1) * TILE_M) * APP_DIM], 0, 0);
    __syncthreads();

    // ---- phase 2a: copy raw 30 features per row (f16) ----
    for (int t = tid; t < TILE_M * 32; t += 256) {
      int m = t >> 5, c = t & 31;
      if (c < NCH) s_in[m * KPAD + c] = (_Float16)s_raw[t];
    }
    // ---- phase 2b: PE, channel-major: 12 contiguous sin/cos per channel ----
    for (int task = tid; task < TILE_M * NCH; task += 256) {
      int m = task & 15, ch = task >> 4;          // 16 rows x 30 channels
      float x = s_raw[m * 32 + ch];
      _Float16 vals[12];
      float xf = x;
#pragma unroll
      for (int f = 0; f < 6; ++f) {
        vals[f]     = (_Float16)__sinf(xf);
        vals[6 + f] = (_Float16)__cosf(xf);
        xf *= 2.0f;
      }
      half2v* dst = (half2v*)&s_in[m * KPAD + 30 + ch * 12];  // 4B aligned
#pragma unroll
      for (int i = 0; i < 6; ++i) {
        half2v pr; pr[0] = vals[2 * i]; pr[1] = vals[2 * i + 1];
        dst[i] = pr;
      }
    }
    __syncthreads();

    // ---- phase 3: layer 1, 13 WMMAs, 2-deep A-fragment pipeline ----
    v8f acc;
#pragma unroll
    for (int r = 0; r < 8; ++r) acc[r] = bv1;
    {
      v16h acur = load_a_lds(s_in, KPAD, n, 0, kHalf);
#pragma unroll
      for (int kt = 0; kt < 13; ++kt) {
        v16h anxt = acur;
        if (kt + 1 < 13) anxt = load_a_lds(s_in, KPAD, n, (kt + 1) * 32, kHalf);
        acc = __builtin_amdgcn_wmma_f32_16x16x32_f16(false, acur, false, w1f[kt],
                                                     (short)0, acc, false, false);
        acur = anxt;
      }
    }
#pragma unroll
    for (int r = 0; r < 8; ++r) {
      float v = fmaxf(acc[r], 0.0f);
      s_h1[(r + 8 * kHalf) * FEAT + 16 * wv + n] = (_Float16)v;
    }
    __syncthreads();

    // ---- phase 4: layer 2, 4 WMMAs ----
#pragma unroll
    for (int r = 0; r < 8; ++r) acc[r] = bv2;
    {
      v16h acur = load_a_lds(s_h1, FEAT, n, 0, kHalf);
#pragma unroll
      for (int kt = 0; kt < 4; ++kt) {
        v16h anxt = acur;
        if (kt + 1 < 4) anxt = load_a_lds(s_h1, FEAT, n, (kt + 1) * 32, kHalf);
        acc = __builtin_amdgcn_wmma_f32_16x16x32_f16(false, acur, false, w2f[kt],
                                                     (short)0, acc, false, false);
        acur = anxt;
      }
    }
#pragma unroll
    for (int r = 0; r < 8; ++r) {
      float v = fmaxf(acc[r], 0.0f);
      s_h2[(r + 8 * kHalf) * FEAT + 16 * wv + n] = (_Float16)v;
    }
    __syncthreads();

    // ---- phase 5: rgb head + weighted reduction (wave 0; scalar-uniform) ----
    if (wv == 0) {
      v8f racc;
#pragma unroll
      for (int r = 0; r < 8; ++r) racc[r] = bv3;
      v16h acur = load_a_lds(s_h2, FEAT, n, 0, kHalf);
#pragma unroll
      for (int kt = 0; kt < 4; ++kt) {
        v16h anxt = acur;
        if (kt + 1 < 4) anxt = load_a_lds(s_h2, FEAT, n, (kt + 1) * 32, kHalf);
        racc = __builtin_amdgcn_wmma_f32_16x16x32_f16(false, acur, false, w3f[kt],
                                                      (short)0, racc, false, false);
        acur = anxt;
      }
      float partial = 0.0f;
#pragma unroll
      for (int r = 0; r < 8; ++r) {
        float sg = 1.0f / (1.0f + __expf(-racc[r]));   // sigmoid
        partial += sg * s_w[r + 8 * kHalf];            // * render weight
      }
      partial += __shfl_xor(partial, 16);              // merge M halves (wave32)
      rgb_sum += partial;
    }
    __syncthreads();
  }

  if (wv == 0 && lane < 3) out[ray * 3 + lane] = rgb_sum;
}

extern "C" void kernel_launch(void* const* d_in, const int* in_sizes, int n_in,
                              void* d_out, int out_size, void* d_ws, size_t ws_size,
                              hipStream_t stream) {
  const float* sigf  = (const float*)d_in[0];
  const float* app   = (const float*)d_in[1];
  const float* vdir  = (const float*)d_in[2];
  const float* dists = (const float*)d_in[3];
  const float* W1    = (const float*)d_in[4];
  const float* b1    = (const float*)d_in[5];
  const float* W2    = (const float*)d_in[6];
  const float* b2    = (const float*)d_in[7];
  const float* W3    = (const float*)d_in[8];
  const float* b3    = (const float*)d_in[9];
  float* out = (float*)d_out;

  const int n_rays = in_sizes[0] / NS;   // 4096
  nerf_fused_kernel<<<n_rays, 256, 0, stream>>>(sigf, app, vdir, dists,
                                                W1, b1, W2, b2, W3, b3, out);
}
